// TemporalGNN_39728447488572
// MI455X (gfx1250) — compile-verified, matrix operand order checked by reading, not run on hardware
//
#include <hip/hip_runtime.h>
#include <hip/hip_bf16.h>
#include <math.h>

#define N_FEAT 8
#define HID    32
#define PER    12
#define XSTR   96   /* F_IN * PERIODS */

typedef __attribute__((ext_vector_type(16))) _Float16 v16h;
typedef __attribute__((ext_vector_type(8)))  _Float16 v8h;
typedef __attribute__((ext_vector_type(8)))  float    v8f;
typedef __attribute__((ext_vector_type(8)))  int      v8i;

// K index mapping for 16-bit A/B operands of v_wmma_*_16x16x32_*:
// element e of the per-lane v16h maps to K = kmap16(e, laneGroup).
__device__ __forceinline__ int kmap16(int e, int g) {
  int v = e >> 1, p = e & 1;
  int base = (v < 4) ? (2 * v) : (16 + 2 * (v - 4));
  return base + (g ? 8 : 0) + p;
}

__device__ __forceinline__ float fast_sigmoid(float x) {
  return __builtin_amdgcn_rcpf(1.0f + __expf(-x));
}
__device__ __forceinline__ float fast_tanh(float x) {
  float e = __expf(-2.0f * x);
  return (1.0f - e) * __builtin_amdgcn_rcpf(1.0f + e);
}

// ---------------- init: zero agg + BN stats, deg = 1 (self loop) ------------
__global__ void tg_init(float* __restrict__ deg, float* __restrict__ agg,
                        float* __restrict__ gstats, int N) {
  long long idx = (long long)blockIdx.x * blockDim.x + threadIdx.x;
  long long tot = (long long)N * XSTR;
  if (idx < tot) agg[idx] = 0.0f;
  if (idx < N)   deg[idx] = 1.0f;     // self-loop weight
  if (idx < 64)  gstats[idx] = 0.0f;  // gsum[32] ++ gsumsq[32]
}

// ---------------- prep: softmax(att), combined weights/biases --------------
// Phase 1 (shared): swc[0:256]=w_z@lz_w[:32], swc[256:512]=w_h@lh_w[:32]
// Phase 2: emit wch = pre-swizzled f16 B operands [gate][coltile][lane][16]
__global__ __launch_bounds__(1024) void tg_prep(
    const float* __restrict__ w_z, const float* __restrict__ b_z,
    const float* __restrict__ w_h, const float* __restrict__ b_h,
    const float* __restrict__ lz_w, const float* __restrict__ lz_b,
    const float* __restrict__ lh_w, const float* __restrict__ lh_b,
    const float* __restrict__ att,
    _Float16* __restrict__ wch, float* __restrict__ bcomb,
    float* __restrict__ probs) {
  __shared__ float swc[512];
  int tid = threadIdx.x;
  if (tid < 512) {
    int sel = tid >> 8, rem = tid & 255, f = rem >> 5, c = rem & 31;
    const float* W = sel ? w_h : w_z;
    const float* L = sel ? lh_w : lz_w;   // rows [0:32) of the [64,32] matrix
    float s = 0.0f;
    for (int j = 0; j < HID; ++j) s += W[f * HID + j] * L[j * HID + c];
    swc[tid] = s;
  } else if (tid < 576) {
    int i = tid - 512, sel = i >> 5, c = i & 31;
    const float* B  = sel ? b_h  : b_z;
    const float* L  = sel ? lh_w : lz_w;
    const float* lb = sel ? lh_b : lz_b;
    float s = lb[c];
    for (int j = 0; j < HID; ++j) s += B[j] * L[j * HID + c];
    bcomb[i] = s;
  } else if (tid == 576) {
    float mx = att[0];
    for (int t = 1; t < PER; ++t) mx = fmaxf(mx, att[t]);
    float ex[PER]; float s = 0.0f;
    for (int t = 0; t < PER; ++t) { ex[t] = __expf(att[t] - mx); s += ex[t]; }
    for (int t = 0; t < PER; ++t) probs[t] = ex[t] / s;
  }
  __syncthreads();
  // 2048 f16 entries: idx = ((gate*2 + ct)*32 + lane)*16 + e
  for (int idx = tid; idx < 2048; idx += 1024) {
    int gate = idx >> 10, rem = idx & 1023;
    int ct = rem >> 9, lane = (rem >> 4) & 31, e = rem & 15;
    int g = lane >> 4, col = ct * 16 + (lane & 15);
    int k = kmap16(e, g);
    float v = (k < N_FEAT) ? swc[gate * 256 + k * HID + col] : 0.0f;
    wch[idx] = (_Float16)v;
  }
}

// ---------------- graph normalization --------------------------------------
__global__ void tg_deg(const long long* __restrict__ ei, float* __restrict__ deg, int E) {
  int e = blockIdx.x * blockDim.x + threadIdx.x;
  if (e < E) atomicAdd(&deg[(size_t)ei[(size_t)E + e]], 1.0f);
}

__global__ void tg_dinv(const float* __restrict__ deg, float* __restrict__ dinv, int N) {
  int i = blockIdx.x * blockDim.x + threadIdx.x;
  if (i < N) dinv[i] = rsqrtf(deg[i]);   // deg >= 1 always (self loop)
}

// ---------------- edge pass: one thread per (edge, element) -----------------
// 96 consecutive threads handle one edge -> coalesced x reads and agg atomics.
__global__ void tg_scatter(const long long* __restrict__ ei,
                           const float* __restrict__ dinv,
                           const float* __restrict__ x,
                           float* __restrict__ agg, int E) {
  long long idx = (long long)blockIdx.x * blockDim.x + threadIdx.x;
  long long tot = (long long)E * XSTR;
  if (idx >= tot) return;
  int e = (int)(idx / XSTR);
  int i = (int)(idx - (long long)e * XSTR);
  long long s = ei[e];
  long long d = ei[(size_t)E + e];
  float nrm = dinv[s] * dinv[d];
  atomicAdd(&agg[(size_t)d * XSTR + i], nrm * x[(size_t)s * XSTR + i]);
}

// ---------------- repack agg (+ self loop) -> f16 WMMA A-operand buffer -----
// aggh[((tile*12 + t)*32 + lane)*8 + f]; lanes 16..31 hold zeros (K padding).
__global__ void tg_repack(const float* __restrict__ agg,
                          const float* __restrict__ dinv,
                          const float* __restrict__ x,
                          _Float16* __restrict__ aggh, int tiles) {
  int idx = blockIdx.x * blockDim.x + threadIdx.x;
  int total = tiles * PER * 32;
  if (idx >= total) return;
  int lane = idx & 31;
  int t    = (idx >> 5) % PER;
  int tile = idx / (PER * 32);
  _Float16* dst = aggh + (size_t)idx * 8;
  if (lane < 16) {
    int n = tile * 16 + lane;
    float w = dinv[n] * dinv[n];           // self-loop weight
    const float* srcA = agg + (size_t)n * XSTR + t;
    const float* srcX = x   + (size_t)n * XSTR + t;
#pragma unroll
    for (int f = 0; f < N_FEAT; ++f)
      dst[f] = (_Float16)(srcA[f * PER] + w * srcX[f * PER]);
  } else {
#pragma unroll
    for (int f = 0; f < N_FEAT; ++f) dst[f] = (_Float16)0.0f;
  }
}

// ---------------- TGCN cell over 12 periods: WMMA f16->f32 -----------------
// One wave32 per 16-node tile. Per period: 1 b128 load, 4 WMMAs, gating.
__global__ __launch_bounds__(32) void tg_cell(
    const _Float16* __restrict__ aggh, const _Float16* __restrict__ wch,
    const float* __restrict__ bcomb, const float* __restrict__ probs,
    _Float16* __restrict__ hbufh, float* __restrict__ gsum,
    float* __restrict__ gsumsq, int N) {
  int lane = threadIdx.x;
  int m = lane & 15, g = lane >> 4;
  int nb = blockIdx.x * 16;

  const v16h* W = (const v16h*)wch;   // [4][32] pre-swizzled B operands
  v16h bz0 = W[0 * 32 + lane];
  v16h bz1 = W[1 * 32 + lane];
  v16h bh0 = W[2 * 32 + lane];
  v16h bh1 = W[3 * 32 + lane];

  float bzc0 = bcomb[m],      bzc1 = bcomb[16 + m];
  float bhc0 = bcomb[32 + m], bhc1 = bcomb[48 + m];

  float acc0[8], acc1[8];
#pragma unroll
  for (int r = 0; r < 8; ++r) { acc0[r] = 0.f; acc1[r] = 0.f; }

  const v8h* A = (const v8h*)aggh;    // 16B per (tile,t,lane)
  size_t abase = (size_t)blockIdx.x * PER * 32 + lane;

#pragma unroll 2
  for (int t = 0; t < PER; ++t) {
    v8h lo = A[abase + (size_t)t * 32];
    v16h a = {};                      // K=16..31 stay zero
#pragma unroll
    for (int i = 0; i < 8; ++i) a[i] = lo[i];

    v8f c0 = {};
    v8f z0 = __builtin_amdgcn_wmma_f32_16x16x32_f16(false, a, false, bz0, (short)0, c0, false, false);
    v8f z1 = __builtin_amdgcn_wmma_f32_16x16x32_f16(false, a, false, bz1, (short)0, c0, false, false);
    v8f h0 = __builtin_amdgcn_wmma_f32_16x16x32_f16(false, a, false, bh0, (short)0, c0, false, false);
    v8f h1 = __builtin_amdgcn_wmma_f32_16x16x32_f16(false, a, false, bh1, (short)0, c0, false, false);
    float p = probs[t];               // uniform scalar load per period
#pragma unroll
    for (int r = 0; r < 8; ++r) {
      float Z0 = fast_sigmoid(z0[r] + bzc0);
      float Z1 = fast_sigmoid(z1[r] + bzc1);
      float H0 = fast_tanh(h0[r] + bhc0);
      float H1 = fast_tanh(h1[r] + bhc1);
      acc0[r] += p * (1.f - Z0) * H0;
      acc1[r] += p * (1.f - Z1) * H1;
    }
  }

  // leaky ReLU, stash pre-BN activations (f16), accumulate BN stats (f32)
  float s0 = 0.f, s1 = 0.f, q0 = 0.f, q1 = 0.f;
#pragma unroll
  for (int r = 0; r < 8; ++r) {
    int row = r + 8 * g;  // C/D layout: lanes 16-31 hold rows 8..15
    float h = acc0[r]; h = h > 0.f ? h : 0.01f * h;
    hbufh[(size_t)(nb + row) * HID + m] = (_Float16)h; s0 += h; q0 += h * h;
    float h2 = acc1[r]; h2 = h2 > 0.f ? h2 : 0.01f * h2;
    hbufh[(size_t)(nb + row) * HID + 16 + m] = (_Float16)h2; s1 += h2; q1 += h2 * h2;
  }
  atomicAdd(&gsum[m], s0);        atomicAdd(&gsumsq[m], q0);
  atomicAdd(&gsum[16 + m], s1);   atomicAdd(&gsumsq[16 + m], q1);
}

// ---------------- fold BN affine into the output linear --------------------
// Bouth = pre-swizzled f16 B operand of scale∘lin_w (K=32, cols 12..15 zero)
// biasOut[p] = sum_k shift[k]*lin_w[k][p] + lin_b[p]
__global__ __launch_bounds__(32) void tg_bnprep(
    const float* __restrict__ gsum, const float* __restrict__ gsumsq,
    const float* __restrict__ gamma, const float* __restrict__ beta,
    const float* __restrict__ lin_w, const float* __restrict__ lin_b,
    _Float16* __restrict__ Bouth, float* __restrict__ biasOut, int N) {
  __shared__ float sscale[HID];
  __shared__ float sshift[HID];
  int c = threadIdx.x;
  float invN = 1.0f / (float)N;
  float mean = gsum[c] * invN;
  float var  = gsumsq[c] * invN - mean * mean;
  float sc = gamma[c] * rsqrtf(var + 1e-5f);
  float sh = beta[c] - mean * sc;
  sscale[c] = sc; sshift[c] = sh;
  __syncthreads();
  int g = c >> 4, p = c & 15;
#pragma unroll
  for (int e = 0; e < 16; ++e) {
    int k = kmap16(e, g);
    float v = (p < PER) ? sscale[k] * lin_w[k * PER + p] : 0.0f;
    Bouth[c * 16 + e] = (_Float16)v;
  }
  if (c < 16) {
    float b = 0.0f;
    if (c < PER) {
      b = lin_b[c];
      for (int k = 0; k < HID; ++k) b += sshift[k] * lin_w[k * PER + c];
    }
    biasOut[c] = b;
  }
}

// ---------------- output head: [16,32] @ [32,16] WMMA, write cols < 12 ------
__global__ __launch_bounds__(32) void tg_out(
    const _Float16* __restrict__ hbufh, const _Float16* __restrict__ Bouth,
    const float* __restrict__ biasOut, float* __restrict__ out, int N) {
  int lane = threadIdx.x;
  int m = lane & 15, g = lane >> 4;
  int nb = blockIdx.x * 16;

  // A operand: row m of hbufh, K pairs are consecutive -> 8 dword loads
  const _Float16* hp = hbufh + (size_t)(nb + m) * HID;
  union { v8i i; v16h h; } au;
#pragma unroll
  for (int v = 0; v < 8; ++v) {
    int k0 = ((v < 4) ? (2 * v) : (16 + 2 * (v - 4))) + (g ? 8 : 0);
    au.i[v] = *(const int*)(hp + k0);
  }
  v16h b = ((const v16h*)Bouth)[lane];

  v8f c0 = {};
  v8f d = __builtin_amdgcn_wmma_f32_16x16x32_f16(false, au.h, false, b, (short)0, c0, false, false);
  if (m < PER) {
    float bb = biasOut[m];
#pragma unroll
    for (int r = 0; r < 8; ++r) {
      int row = r + 8 * g;
      out[(size_t)(nb + row) * PER + m] = d[r] + bb;
    }
  }
}

extern "C" void kernel_launch(void* const* d_in, const int* in_sizes, int n_in,
                              void* d_out, int out_size, void* d_ws, size_t ws_size,
                              hipStream_t stream) {
  const float*     x     = (const float*)d_in[0];
  const long long* ei    = (const long long*)d_in[1];   // int64 edge_index [2,E]
  const float*     w_z   = (const float*)d_in[2];
  const float*     b_z   = (const float*)d_in[3];
  const float*     w_h   = (const float*)d_in[6];
  const float*     b_h   = (const float*)d_in[7];
  const float*     lz_w  = (const float*)d_in[8];
  const float*     lz_b  = (const float*)d_in[9];
  const float*     lh_w  = (const float*)d_in[12];
  const float*     lh_b  = (const float*)d_in[13];
  const float*     att   = (const float*)d_in[14];
  const float*     gamma = (const float*)d_in[15];
  const float*     beta  = (const float*)d_in[16];
  const float*     lin_w = (const float*)d_in[17];
  const float*     lin_b = (const float*)d_in[18];
  float*           out   = (float*)d_out;

  const int N = in_sizes[0] / XSTR;   // 50000
  const int E = in_sizes[1] / 2;      // 1600000
  const int tiles = N / 16;           // 3125 exactly

  float* ws = (float*)d_ws;
  size_t off = 0;
  float* agg      = ws + off; off += (size_t)N * XSTR;        // f32 atomics target
  float* deg      = ws + off; off += (size_t)N;
  float* dinv     = ws + off; off += (size_t)N;
  float* gsum     = ws + off; off += 32;
  float* gsumsq   = ws + off; off += 32;    // contiguous 64-float stats block
  float* bcomb    = ws + off; off += 64;
  float* probs    = ws + off; off += 16;
  float* biasOut  = ws + off; off += 16;
  _Float16* wch   = (_Float16*)(ws + off); off += 1024;       // 2048 halves
  _Float16* Bouth = (_Float16*)(ws + off); off += 256;        // 512 halves
  _Float16* aggh  = (_Float16*)(ws + off); off += (size_t)N * XSTR; // N*192 halves
  _Float16* hbufh = (_Float16*)(ws + off); off += (size_t)N * 16;   // N*32 halves

  const int THR = 256;
  long long totAgg  = (long long)N * XSTR;
  long long totScat = (long long)E * XSTR;
  int gAgg  = (int)((totAgg + THR - 1) / THR);
  int gScat = (int)((totScat + THR - 1) / THR);
  int gEdge = (E + THR - 1) / THR;
  int gNode = (N + THR - 1) / THR;
  int gRep  = (tiles * PER * 32 + THR - 1) / THR;

  tg_init    <<<gAgg,  THR,  0, stream>>>(deg, agg, gsum, N);
  tg_prep    <<<1,     1024, 0, stream>>>(w_z, b_z, w_h, b_h, lz_w, lz_b, lh_w, lh_b,
                                          att, wch, bcomb, probs);
  tg_deg     <<<gEdge, THR,  0, stream>>>(ei, deg, E);
  tg_dinv    <<<gNode, THR,  0, stream>>>(deg, dinv, N);
  tg_scatter <<<gScat, THR,  0, stream>>>(ei, dinv, x, agg, E);
  tg_repack  <<<gRep,  THR,  0, stream>>>(agg, dinv, x, aggh, tiles);
  tg_cell    <<<tiles, 32,   0, stream>>>(aggh, wch, bcomb, probs, hbufh,
                                          gsum, gsumsq, N);
  tg_bnprep  <<<1,     32,   0, stream>>>(gsum, gsumsq, gamma, beta, lin_w, lin_b,
                                          Bouth, biasOut, N);
  tg_out     <<<tiles, 32,   0, stream>>>(hbufh, Bouth, biasOut, out, N);
}
